// GroupedQueryAttention_38920993636956
// MI455X (gfx1250) — matrix operand order repaired and checked by reading, not compiled, not run on hardware
//
#include <hip/hip_runtime.h>

// ---------------------------------------------------------------- types
typedef __bf16 bf16;
typedef __attribute__((ext_vector_type(16))) __bf16 bf16x16;
typedef __attribute__((ext_vector_type(8)))  __bf16 bf16x8;
typedef __attribute__((ext_vector_type(8)))  float  v8f;

#define BB     2
#define TT     2048
#define DMODEL 2048
#define HH     32
#define KVH    8
#define HDIM   64
#define GG     4
#define DKV    (KVH * HDIM)   // 512
#define BT     (BB * TT)      // 4096

// ---------------------------------------------------------------- helpers
// A fragment (16x32 bf16, M x K): lanes 0-15 -> M=lane, K=k0..k0+15
//                                 lanes 16-31 -> M=lane-16, K=k0+16..k0+31
__device__ __forceinline__ bf16x16 load_a_frag(const bf16* __restrict__ base, int ld,
                                               int m0, int k0, int lane) {
  int m = m0 + (lane & 15);
  int k = k0 + ((lane >> 4) ? 16 : 0);
  union { bf16x16 v; bf16x8 h[2]; } u;
  u.h[0] = *reinterpret_cast<const bf16x8*>(base + (size_t)m * ld + k);
  u.h[1] = *reinterpret_cast<const bf16x8*>(base + (size_t)m * ld + k + 8);
  return u.v;
}

// B fragment (32x16 bf16, K x N), sourced from row-major W[n, k] (i.e. B = W^T):
// lanes 0-15 -> N=lane, K=k0..k0+15 ; lanes 16-31 -> N=lane-16, K=k0+16..k0+31
__device__ __forceinline__ bf16x16 load_b_frag(const bf16* __restrict__ base, int ld,
                                               int n0, int k0, int lane) {
  int n = n0 + (lane & 15);
  int k = k0 + ((lane >> 4) ? 16 : 0);
  return *reinterpret_cast<const bf16x16*>(base + (size_t)n * ld + k);
}

__device__ __forceinline__ v8f wmma_bf16(bf16x16 a, bf16x16 b, v8f c) {
  return __builtin_amdgcn_wmma_f32_16x16x32_bf16(false, a, false, b, (short)0, c,
                                                 false, false);
}

// ---------------------------------------------------------------- convert
__global__ void f32_to_bf16_kernel(const float* __restrict__ in,
                                   bf16* __restrict__ out, int n) {
  int i = blockIdx.x * blockDim.x + threadIdx.x;
  if (i < n) out[i] = (bf16)in[i];
}

// ---------------------------------------------------------------- GEMM: C = A * W^T
// A: MxK row-major bf16, W: NxK row-major bf16.
// Block tile 128(M) x 256(N); 8 waves as 2(M) x 4(N); 64x64 per wave (4x4 WMMA tiles)
// -> 16 WMMAs per 8 fragment loads per K-step (1.5x the FLOP/B of a 32x64 wave tile).
// OutT: bf16 or float. TRANS=true stores C transposed (C[n*ldc+m]).
template <typename OutT, bool TRANS>
__global__ __launch_bounds__(256) void gemm_bf16_wmma(
    const bf16* __restrict__ A, const bf16* __restrict__ W, OutT* __restrict__ C,
    int M, int N, int K, int ldc) {
  (void)M; (void)N;
  int lane = threadIdx.x & 31;
  int wave = threadIdx.x >> 5;
  int bm = blockIdx.y * 128 + (wave & 1) * 64;   // 2 waves along M
  int bn = blockIdx.x * 256 + (wave >> 1) * 64;  // 4 waves along N

  v8f acc[4][4];
#pragma unroll
  for (int i = 0; i < 4; ++i)
#pragma unroll
    for (int j = 0; j < 4; ++j)
#pragma unroll
      for (int e = 0; e < 8; ++e) acc[i][j][e] = 0.0f;

  for (int k0 = 0; k0 < K; k0 += 32) {
    bf16x16 a[4], b[4];
#pragma unroll
    for (int i = 0; i < 4; ++i) a[i] = load_a_frag(A, K, bm + i * 16, k0, lane);
#pragma unroll
    for (int j = 0; j < 4; ++j) b[j] = load_b_frag(W, K, bn + j * 16, k0, lane);

    if (k0 + 32 < K) {  // hint next K-step into cache (global_prefetch_b8)
      __builtin_prefetch(A + (size_t)(bm + (lane & 15)) * K + k0 + 32, 0, 1);
      __builtin_prefetch(W + (size_t)(bn + (lane & 15)) * K + k0 + 32, 0, 1);
    }
#pragma unroll
    for (int i = 0; i < 4; ++i)
#pragma unroll
      for (int j = 0; j < 4; ++j) acc[i][j] = wmma_bf16(a[i], b[j], acc[i][j]);
  }

  int half = lane >> 4, nl = lane & 15;
#pragma unroll
  for (int i = 0; i < 4; ++i)
#pragma unroll
    for (int j = 0; j < 4; ++j)
#pragma unroll
      for (int r = 0; r < 8; ++r) {
        int m = bm + i * 16 + r + 8 * half;
        int n = bn + j * 16 + nl;
        float v = acc[i][j][r];
        if (TRANS) C[(size_t)n * ldc + m] = (OutT)v;
        else       C[(size_t)m * ldc + n] = (OutT)v;
      }
}

// ---------------------------------------------------------------- flash attention
// One wave handles one 16-row query tile of one head. Causal GQA.
// Q:  [BT, DMODEL] bf16 (head h at cols h*64..)
// Km: [BT, DKV]    bf16 (kv head at cols kv*64..)
// Vt: [DKV, BT]    bf16 (V transposed)
// O:  [BT, DMODEL] bf16
__global__ __launch_bounds__(256) void gqa_flash_wmma(
    const bf16* __restrict__ Q, const bf16* __restrict__ Km,
    const bf16* __restrict__ Vt, bf16* __restrict__ O) {
  __shared__ bf16 plds[8][16 * 32];

  int lane = threadIdx.x & 31;
  int wave = threadIdx.x >> 5;
  int tile = blockIdx.x * 8 + wave;   // grid sized exactly: no bounds check
  const int ntq = TT / 16;            // 128 query tiles per head
  int qt = tile % ntq;
  int h  = (tile / ntq) % HH;
  int b  = tile / (ntq * HH);
  int kv = h / GG;
  int q0 = qt * 16;
  int half = lane >> 4, nl = lane & 15;

  // Preload Q A-fragments for the two 32-wide K-dim steps of HD=64.
  bf16x16 qa[2];
#pragma unroll
  for (int d = 0; d < 2; ++d) {
    size_t row = (size_t)(b * TT + q0 + nl);
    int k = h * HDIM + d * 32 + (half ? 16 : 0);
    union { bf16x16 v; bf16x8 hv[2]; } u;
    u.hv[0] = *reinterpret_cast<const bf16x8*>(Q + row * DMODEL + k);
    u.hv[1] = *reinterpret_cast<const bf16x8*>(Q + row * DMODEL + k + 8);
    qa[d] = u.v;
  }

  v8f o[4];
#pragma unroll
  for (int j = 0; j < 4; ++j)
#pragma unroll
    for (int e = 0; e < 8; ++e) o[j][e] = 0.0f;

  float mstat[8], lstat[8];
#pragma unroll
  for (int r = 0; r < 8; ++r) { mstat[r] = -3.0e38f; lstat[r] = 0.0f; }

  for (int s0 = 0; s0 < q0 + 16; s0 += 32) {   // 32 keys per chunk
    // ---- scores S = (Q K^T) * scale, causal-masked. Two 16-key sub-chunks.
    v8f sf[2];
#pragma unroll
    for (int nc = 0; nc < 2; ++nc) {
      int sn = s0 + nc * 16;
      v8f s;
#pragma unroll
      for (int e = 0; e < 8; ++e) s[e] = 0.0f;
      int key = sn + nl;
#pragma unroll
      for (int d = 0; d < 2; ++d) {
        int k = kv * HDIM + d * 32 + (half ? 16 : 0);
        bf16x16 bk = *reinterpret_cast<const bf16x16*>(
            Km + (size_t)(b * TT + key) * DKV + k);
        s = wmma_bf16(qa[d], bk, s);
      }
#pragma unroll
      for (int r = 0; r < 8; ++r) {
        int mq = q0 + r + 8 * half;
        float sv = s[r] * 0.125f;                 // 1/sqrt(64)
        sf[nc][r] = (key > mq) ? -3.0e38f : sv;   // causal mask
      }
    }

    // ---- online softmax (row stats; rows live across 16 lanes of a half)
    float p0[8], p1[8];
#pragma unroll
    for (int r = 0; r < 8; ++r) {
      float cm = fmaxf(sf[0][r], sf[1][r]);
#pragma unroll
      for (int msk = 1; msk <= 8; msk <<= 1)
        cm = fmaxf(cm, __shfl_xor(cm, msk, 32));
      float mold = mstat[r];
      float mnew = fmaxf(mold, cm);
      float alpha = __expf(mold - mnew);
      float e0 = __expf(sf[0][r] - mnew);
      float e1 = __expf(sf[1][r] - mnew);
      float ps = e0 + e1;
#pragma unroll
      for (int msk = 1; msk <= 8; msk <<= 1) ps += __shfl_xor(ps, msk, 32);
      lstat[r] = lstat[r] * alpha + ps;
      mstat[r] = mnew;
      p0[r] = e0; p1[r] = e1;
#pragma unroll
      for (int j = 0; j < 4; ++j) o[j][r] *= alpha;
    }

    // ---- re-layout P (D-fragment) -> A-fragment via per-wave LDS tile
    bf16* pw = plds[wave];
#pragma unroll
    for (int r = 0; r < 8; ++r) {
      int m = r + 8 * half;
      pw[m * 32 + nl]      = (bf16)p0[r];
      pw[m * 32 + 16 + nl] = (bf16)p1[r];
    }
    asm volatile("s_wait_dscnt 0" ::: "memory");  // wave-internal LDS visibility
    union { bf16x16 v; bf16x8 hv[2]; } up;
    {
      int m = nl;
      int ka = half ? 16 : 0;
      up.hv[0] = *reinterpret_cast<const bf16x8*>(pw + m * 32 + ka);
      up.hv[1] = *reinterpret_cast<const bf16x8*>(pw + m * 32 + ka + 8);
    }

    // ---- O += P * V  (V^T rows give contiguous 16-key B-fragments)
#pragma unroll
    for (int j = 0; j < 4; ++j) {
      int dcol = kv * HDIM + j * 16 + nl;
      int key  = b * TT + s0 + (half ? 16 : 0);
      bf16x16 bv = *reinterpret_cast<const bf16x16*>(Vt + (size_t)dcol * BT + key);
      o[j] = wmma_bf16(up.v, bv, o[j]);
    }
  }

  // ---- normalize and store
#pragma unroll
  for (int r = 0; r < 8; ++r) {
    float inv = 1.0f / lstat[r];
    size_t row = (size_t)(b * TT + q0 + r + 8 * half);
#pragma unroll
    for (int j = 0; j < 4; ++j)
      O[row * DMODEL + h * HDIM + j * 16 + nl] = (bf16)(o[j][r] * inv);
  }
}

// ---------------------------------------------------------------- launcher
extern "C" void kernel_launch(void* const* d_in, const int* in_sizes, int n_in,
                              void* d_out, int out_size, void* d_ws, size_t ws_size,
                              hipStream_t stream) {
  (void)in_sizes; (void)n_in; (void)out_size; (void)ws_size;
  const float* x  = (const float*)d_in[0];
  const float* Wq = (const float*)d_in[1];
  const float* Wk = (const float*)d_in[2];
  const float* Wv = (const float*)d_in[3];
  const float* Wo = (const float*)d_in[4];
  // d_in[5] (mask) is ignored: causal mask applied analytically.
  float* out = (float*)d_out;

  char* ws = (char*)d_ws;
  size_t off = 0;
  auto take = [&](size_t nelem) -> bf16* {
    bf16* p = (bf16*)(ws + off);
    off += (nelem * sizeof(bf16) + 255) & ~(size_t)255;
    return p;
  };
  bf16* xb  = take((size_t)BT * DMODEL);     // x bf16
  bf16* wqb = take((size_t)DMODEL * DMODEL); // Wq bf16
  bf16* wkb = take((size_t)DKV * DMODEL);    // Wk bf16
  bf16* wvb = take((size_t)DKV * DMODEL);    // Wv bf16
  bf16* wob = take((size_t)DMODEL * DMODEL); // Wo bf16
  bf16* qb  = take((size_t)BT * DMODEL);     // Q  [BT, D]
  bf16* kb  = take((size_t)BT * DKV);        // K  [BT, DKV]
  bf16* vtb = take((size_t)DKV * BT);        // V^T [DKV, BT]
  bf16* ob  = take((size_t)BT * DMODEL);     // attention output [BT, D]

  auto cvt = [&](const float* src, bf16* dst, int n) {
    f32_to_bf16_kernel<<<(n + 255) / 256, 256, 0, stream>>>(src, dst, n);
  };
  cvt(x,  xb,  BT * DMODEL);
  cvt(Wq, wqb, DMODEL * DMODEL);
  cvt(Wk, wkb, DKV * DMODEL);
  cvt(Wv, wvb, DKV * DMODEL);
  cvt(Wo, wob, DMODEL * DMODEL);

  dim3 blk(256);
  // Q = x Wq^T  (4096 x 2048); block tile 128x256
  gemm_bf16_wmma<bf16, false><<<dim3(DMODEL / 256, BT / 128), blk, 0, stream>>>(
      xb, wqb, qb, BT, DMODEL, DMODEL, DMODEL);
  // K = x Wk^T  (4096 x 512)
  gemm_bf16_wmma<bf16, false><<<dim3(DKV / 256, BT / 128), blk, 0, stream>>>(
      xb, wkb, kb, BT, DKV, DMODEL, DKV);
  // V^T = (x Wv^T)^T  (512 x 4096), stored transposed for PV fragments
  gemm_bf16_wmma<bf16, true><<<dim3(DKV / 256, BT / 128), blk, 0, stream>>>(
      xb, wvb, vtb, BT, DKV, DMODEL, BT);
  // Flash attention: B*H*(T/16) = 8192 wave-tiles, 8 waves per 256-thread block
  gqa_flash_wmma<<<dim3((BB * HH * (TT / 16)) / 8), blk, 0, stream>>>(qb, kb, vtb, ob);
  // out = attn Wo^T  (4096 x 2048), f32 output
  gemm_bf16_wmma<float, false><<<dim3(DMODEL / 256, BT / 128), blk, 0, stream>>>(
      ob, wob, out, BT, DMODEL, DMODEL, DMODEL);
}
